// NeuralCodebook_54889682043179
// MI455X (gfx1250) — matrix operand order repaired and checked by reading, not compiled, not run on hardware
//
#include <hip/hip_runtime.h>
#include <hip/hip_bf16.h>
#include <math.h>

// Problem constants (B=4, N=4096, D=64, K=8192)
#define BN_   16384
#define D_    64
#define K_    8192
#define EPSN  1e-12f
#define NTILE (K_ / 16)                // 512 code tiles of 16
#define GL_BLOCKS ((BN_ * D_) / 256)   // 4096 gather/loss blocks

typedef __attribute__((ext_vector_type(16))) __bf16 v16bf;
typedef __attribute__((ext_vector_type(8)))  float  v8f;

__device__ __forceinline__ unsigned short f2bf(float f) {
    // round-to-nearest-even f32 -> bf16
    unsigned int u = __float_as_uint(f);
    u += 0x7FFFu + ((u >> 16) & 1u);
    return (unsigned short)(u >> 16);
}

// ---------------------------------------------------------------------------
// Kernel 1: row-wise L2 normalize, emit f32 and bf16 copies.
// One wave (32 lanes) per row of 64 floats; lane handles a float2.
// ---------------------------------------------------------------------------
__global__ void nc_normalize(const float* __restrict__ src,
                             float* __restrict__ dstF,
                             unsigned short* __restrict__ dstB,
                             int nrows) {
    int wave = (int)((blockIdx.x * blockDim.x + threadIdx.x) >> 5);
    int lane = threadIdx.x & 31;
    if (wave >= nrows) return;
    const float2 v = ((const float2*)(src + (size_t)wave * D_))[lane];
    float ss = v.x * v.x + v.y * v.y;
    #pragma unroll
    for (int m = 16; m >= 1; m >>= 1) ss += __shfl_xor(ss, m, 32);
    float inv = 1.0f / fmaxf(sqrtf(ss), EPSN);
    float x = v.x * inv, y = v.y * inv;
    ((float2*)(dstF + (size_t)wave * D_))[lane] = make_float2(x, y);
    unsigned int packed = (unsigned int)f2bf(x) | ((unsigned int)f2bf(y) << 16);
    ((unsigned int*)(dstB + (size_t)wave * D_))[lane] = packed;
}

// ---------------------------------------------------------------------------
// Kernel 2: fused similarity-GEMM + argmax via v_wmma_f32_16x16x32_bf16.
// One wave per 16-row z tile; streams all 8192 codes (L2-resident, 1 MB).
// A fragments loaded once, reused for all tiles. Two independent code tiles
// per iteration (2 accumulators) to fill WMMA->WMMA RAW hazard slots with
// independent matrix ops instead of v_nops; unconditional prefetch keeps the
// loop a single straight-line block.
// ---------------------------------------------------------------------------
__global__ __launch_bounds__(128) void nc_argmax_wmma(
        const unsigned short* __restrict__ zb,   // [BN_, D_] bf16 bits (normalized z)
        const unsigned short* __restrict__ eb,   // [K_,  D_] bf16 bits (normalized codes)
        int* __restrict__ idx_out) {
    const int warp = threadIdx.x >> 5;
    const int lane = threadIdx.x & 31;
    const int rowTile = blockIdx.x * 4 + warp;   // 0..1023
    const int rowBase = rowTile * 16;
    const int m  = lane & 15;
    const int hi = lane >> 4;

    // A fragments (contiguous K order, matching B):
    // halves 0..15 = K (hi*16 .. hi*16+15)
    const unsigned short* zr = zb + (size_t)(rowBase + m) * D_;
    union { uint4 u[2]; v16bf v; } a0u, a1u;
    a0u.u[0] = *(const uint4*)(zr + hi * 16);
    a0u.u[1] = *(const uint4*)(zr + hi * 16 + 8);
    a1u.u[0] = *(const uint4*)(zr + 32 + hi * 16);
    a1u.u[1] = *(const uint4*)(zr + 40 + hi * 16);

    float bestv[8];
    int   besti[8];
    #pragma unroll
    for (int r = 0; r < 8; ++r) { bestv[r] = -3.0e38f; besti[r] = 0; }

    for (int t = 0; t < NTILE / 2; ++t) {
        const int c0 = t * 32;
        // B fragments: lane n holds column (code) c0+n, contiguous 16 features;
        // lane half selects K-subrange (0-15 vs 16-31) in contiguous order, matching A.
        const unsigned short* er0 = eb + (size_t)(c0 + m) * D_ + hi * 16;
        const unsigned short* er1 = er0 + 16 * D_;
        union { uint4 u[2]; v16bf v; } b0a, b0b, b1a, b1b;
        b0a.u[0] = *(const uint4*)(er0);
        b0a.u[1] = *(const uint4*)(er0 + 8);
        b0b.u[0] = *(const uint4*)(er0 + 32);   // features 32..63 chunk
        b0b.u[1] = *(const uint4*)(er0 + 40);
        b1a.u[0] = *(const uint4*)(er1);
        b1a.u[1] = *(const uint4*)(er1 + 8);
        b1b.u[0] = *(const uint4*)(er1 + 32);
        b1b.u[1] = *(const uint4*)(er1 + 40);
        // Unconditional prefetch of the next 32-code chunk: keeps the loop a
        // single block; the one-past-the-end prefetch hits valid ws memory.
        __builtin_prefetch((const void*)(er0 + 32 * D_), 0, 1);

        v8f acc0 = {}, acc1 = {};
        // Interleave independent tiles so dependent WMMAs are never adjacent.
        acc0 = __builtin_amdgcn_wmma_f32_16x16x32_bf16(
                false, a0u.v, false, b0a.v, (short)0, acc0, false, false);
        acc1 = __builtin_amdgcn_wmma_f32_16x16x32_bf16(
                false, a0u.v, false, b1a.v, (short)0, acc1, false, false);
        acc0 = __builtin_amdgcn_wmma_f32_16x16x32_bf16(
                false, a1u.v, false, b0b.v, (short)0, acc0, false, false);
        acc1 = __builtin_amdgcn_wmma_f32_16x16x32_bf16(
                false, a1u.v, false, b1b.v, (short)0, acc1, false, false);

        const int code0 = c0 + m;         // tile 0 codes (smaller indices first:
        const int code1 = c0 + 16 + m;    //  preserves argmin first-index ties)
        #pragma unroll
        for (int r = 0; r < 8; ++r) {
            float v0 = acc0[r];
            bool u0 = v0 > bestv[r];
            bestv[r] = u0 ? v0 : bestv[r];
            besti[r] = u0 ? code0 : besti[r];
            float v1 = acc1[r];
            bool u1 = v1 > bestv[r];
            bestv[r] = u1 ? v1 : bestv[r];
            besti[r] = u1 ? code1 : besti[r];
        }
    }

    // Cross-lane argmax within each 16-lane half (rows 0-7 low half, 8-15 high).
    #pragma unroll
    for (int r = 0; r < 8; ++r) {
        float bv = bestv[r]; int bi = besti[r];
        #pragma unroll
        for (int msk = 8; msk >= 1; msk >>= 1) {
            float ov = __shfl_xor(bv, msk, 32);
            int   oi = __shfl_xor(bi, msk, 32);
            if (ov > bv || (ov == bv && oi < bi)) { bv = ov; bi = oi; }
        }
        bestv[r] = bv; besti[r] = bi;
    }
    if ((lane & 15) == 0) {
        int base = rowBase + hi * 8;
        #pragma unroll
        for (int r = 0; r < 8; ++r) idx_out[base + r] = besti[r];
    }
}

// ---------------------------------------------------------------------------
// Kernel 3: gather normalized codes, write quantized output + indices (as f32),
// and per-block squared-error partial sums (deterministic, no float atomics).
// ---------------------------------------------------------------------------
__global__ void nc_gather_loss(const int* __restrict__ idx,
                               const float* __restrict__ embN,
                               const float* __restrict__ zN,
                               float* __restrict__ outQ,
                               float* __restrict__ outIdx,
                               float* __restrict__ partial) {
    __shared__ float red[256];
    int i   = blockIdx.x * 256 + threadIdx.x;
    int row = i >> 6;
    int col = i & 63;
    int k = idx[row];
    float q = embN[(size_t)k * D_ + col];
    outQ[i] = q;
    if (col == 0) outIdx[row] = (float)k;
    float d = zN[i] - q;
    red[threadIdx.x] = d * d;
    __syncthreads();
    #pragma unroll
    for (int s = 128; s >= 1; s >>= 1) {
        if (threadIdx.x < s) red[threadIdx.x] += red[threadIdx.x + s];
        __syncthreads();
    }
    if (threadIdx.x == 0) partial[blockIdx.x] = red[0];
}

// ---------------------------------------------------------------------------
// Kernel 4: fixed-order reduction of partials -> loss = 2 * mean (BETA = 1).
// ---------------------------------------------------------------------------
__global__ void nc_finalize(const float* __restrict__ partial,
                            float* __restrict__ outLoss) {
    __shared__ float red[256];
    float s = 0.0f;
    for (int i = threadIdx.x; i < GL_BLOCKS; i += 256) s += partial[i];
    red[threadIdx.x] = s;
    __syncthreads();
    #pragma unroll
    for (int t = 128; t >= 1; t >>= 1) {
        if (threadIdx.x < t) red[threadIdx.x] += red[threadIdx.x + t];
        __syncthreads();
    }
    if (threadIdx.x == 0)
        outLoss[0] = red[0] * (2.0f / (float)(BN_ * D_));
}

// ---------------------------------------------------------------------------
extern "C" void kernel_launch(void* const* d_in, const int* in_sizes, int n_in,
                              void* d_out, int out_size, void* d_ws, size_t ws_size,
                              hipStream_t stream) {
    const float* z   = (const float*)d_in[0];   // [4,4096,64]
    const float* emb = (const float*)d_in[1];   // [8192,64]

    char* ws = (char*)d_ws;
    size_t off = 0;
    float*          embN_f32 = (float*)(ws + off);          off += (size_t)K_  * D_ * 4;  // 2 MB
    unsigned short* embN_bf  = (unsigned short*)(ws + off); off += (size_t)K_  * D_ * 2;  // 1 MB
    float*          zN_f32   = (float*)(ws + off);          off += (size_t)BN_ * D_ * 4;  // 4 MB
    unsigned short* zN_bf    = (unsigned short*)(ws + off); off += (size_t)BN_ * D_ * 2;  // 2 MB
    int*            idx      = (int*)(ws + off);            off += (size_t)BN_ * 4;       // 64 KB
    float*          partial  = (float*)(ws + off);          off += (size_t)GL_BLOCKS * 4; // 16 KB

    float* outQ    = (float*)d_out;              // [BN_, D_]
    float* outIdx  = outQ + (size_t)BN_ * D_;    // [BN_]
    float* outLoss = outIdx + BN_;               // [1]

    nc_normalize<<<K_  / 8, 256, 0, stream>>>(emb, embN_f32, embN_bf, K_);
    nc_normalize<<<BN_ / 8, 256, 0, stream>>>(z, zN_f32, zN_bf, BN_);
    nc_argmax_wmma<<<BN_ / 64, 128, 0, stream>>>(zN_bf, embN_bf, idx);
    nc_gather_loss<<<GL_BLOCKS, 256, 0, stream>>>(idx, embN_f32, zN_f32, outQ, outIdx, partial);
    nc_finalize<<<1, 256, 0, stream>>>(partial, outLoss);
}
